// PermutationROUNDModel_42691974922372
// MI455X (gfx1250) — matrix operand order repaired and checked by reading, not compiled, not run on hardware
//
#include <hip/hip_runtime.h>

// ---------------------------------------------------------------------------
// Types for CDNA5 WMMA (wave32)
// ---------------------------------------------------------------------------
typedef __bf16 v16bf __attribute__((ext_vector_type(16)));
typedef float  v8f   __attribute__((ext_vector_type(8)));

union BF16Frag {
    v16bf v;
    unsigned short u[16];
    uint4 q[2];
};

#define TWO_PI_F     6.28318530717958647692f
#define FOUR_PI_F    12.5663706143591729539f
#define INV_TWO_PI_F 0.15915494309189533577f

// Problem constants (reference: B=16,S=2048,D=128,H=512,O=1024)
#define MROWS 32768   // B*S
#define DDIM  128
#define HDIM  512
#define FDIM  2560    // 5*H
#define ODIM  1024
#define KCHUNK 64
#define NKCHUNK (FDIM / KCHUNK)   // 40

__device__ __forceinline__ unsigned short f2bf(float f) {
    unsigned int b = __float_as_uint(f);
    unsigned int r = b + 0x7FFFu + ((b >> 16) & 1u);   // round-to-nearest-even
    return (unsigned short)(r >> 16);
}

// Issue one 32-byte async stage: two b128 lane-transfers; the instruction
// offset is added to BOTH the LDS and global addresses (ISA 10.x async ops).
__device__ __forceinline__ void async_stage32(unsigned ldsaddr,
                                              const unsigned short* g) {
    asm volatile("global_load_async_to_lds_b128 %0, %1, off"
                 :: "v"(ldsaddr), "v"(g) : "memory");
    asm volatile("global_load_async_to_lds_b128 %0, %1, off offset:16"
                 :: "v"(ldsaddr), "v"(g) : "memory");
}

// ---------------------------------------------------------------------------
// Kernel 0: f32 -> bf16 conversion (weights + activations, done once)
// ---------------------------------------------------------------------------
__global__ void cvt_bf16_kernel(const float* __restrict__ in,
                                unsigned short* __restrict__ out, int n) {
    int i = blockIdx.x * blockDim.x + threadIdx.x;
    if (i < n) out[i] = f2bf(in[i]);
}

// ---------------------------------------------------------------------------
// Kernel 1: GEMM1 (pt = x * W_e^T + b_e) fused with 2*pi wrap -> delta
//   Wave computes one 16x16 tile over K=128 (4 x K=32 bf16 WMMA).
//   All fragment loads are 16B vector loads (ISA 16-bit A layout:
//   V0..V3 -> K = hi*8+0..7, V4..V7 -> K = 16+hi*8+0..7).
// ---------------------------------------------------------------------------
__global__ void gemm1_delta_kernel(const unsigned short* __restrict__ x16,
                                   const unsigned short* __restrict__ We,
                                   const float* __restrict__ be,
                                   float* __restrict__ delta) {
    const int lane = threadIdx.x & 31;
    const int wave = threadIdx.x >> 5;
    const int tile = blockIdx.x * 4 + wave;
    const int ht = tile & 31;        // 32 H tiles
    const int mt = tile >> 5;        // 2048 M tiles
    const int ln = lane & 15;
    const int hi = lane >> 4;

    const unsigned short* __restrict__ xrow = x16 + (size_t)(mt * 16 + ln) * DDIM;
    const unsigned short* __restrict__ wrow = We + (size_t)(ht * 16 + ln) * DDIM;

    v8f c = {};
#pragma unroll
    for (int k0 = 0; k0 < DDIM; k0 += 32) {
        BF16Frag a, b;
        a.q[0] = *(const uint4*)(xrow + k0 + hi * 8);
        a.q[1] = *(const uint4*)(xrow + k0 + 16 + hi * 8);
        b.q[0] = *(const uint4*)(wrow + k0 + hi * 8);
        b.q[1] = *(const uint4*)(wrow + k0 + 16 + hi * 8);
        c = __builtin_amdgcn_wmma_f32_16x16x32_bf16(false, a.v, false, b.v,
                                                    (short)0, c, false, false);
    }

    const int h = ht * 16 + ln;
    const float bias = be[h];
#pragma unroll
    for (int r = 0; r < 8; ++r) {
        float pt = c[r] + bias;
        // atan2(sin(pt), cos(pt)) == wrap pt into (-pi, pi]
        float d = pt - TWO_PI_F * rintf(pt * INV_TWO_PI_F);
        int row = mt * 16 + hi * 8 + r;
        delta[(size_t)row * HDIM + h] = d;
    }
}

// ---------------------------------------------------------------------------
// Kernel 2: sequential phase scan per (b,h) chain + feature expansion
//   acc in (-2pi,2pi], delta in (-pi,pi] => sum in (-3pi,3pi]: a single
//   conditional +/-4pi replaces fmod in the serial dependence chain.
// ---------------------------------------------------------------------------
__global__ void scan_feats_kernel(const float* __restrict__ delta,
                                  const float* __restrict__ ph_in,
                                  float* __restrict__ hist_out,
                                  unsigned short* __restrict__ feats) {
    int t = blockIdx.x * blockDim.x + threadIdx.x;   // 0..8191
    int b = t >> 9;      // / 512
    int h = t & 511;

    float acc = ph_in[t];
    for (int s = 0; s < 2048; ++s) {
        int row = b * 2048 + s;
        float r = acc + delta[(size_t)row * HDIM + h];
        if (r > TWO_PI_F) r -= FOUR_PI_F;
        if (r <= -TWO_PI_F) r += FOUR_PI_F;
        acc = r;
        float hist = acc;
        hist_out[(size_t)row * HDIM + h] = hist;

        float sh, ch, sh2, ch2;
        __sincosf(hist, &sh, &ch);
        __sincosf(0.5f * hist, &sh2, &ch2);

        unsigned short* f = feats + (size_t)row * FDIM;
        f[h]            = f2bf(ch);
        f[HDIM + h]     = f2bf(sh);
        f[2 * HDIM + h] = f2bf(ch2);
        f[3 * HDIM + h] = f2bf(sh2);
        f[4 * HDIM + h] = f2bf(hist);
    }
}

// ---------------------------------------------------------------------------
// Kernel 3: GEMM2 (logits = feats * W_r^T + b_r), bf16 WMMA, f32 accum.
//   Block (8 waves) computes 128 rows x 64 cols. Each 64x64 bf16 W_r chunk
//   (8KB) is staged into LDS via global_load_async_to_lds_b128 (32B per
//   thread), double-buffered. One barrier per chunk:
//     wait-own-asynccnt -> barrier -> issue next chunk -> compute (8 WMMAs)
//   The barrier jointly proves chunk i landed AND chunk i-1 readers are done.
// ---------------------------------------------------------------------------
__global__ void __launch_bounds__(256)
gemm2_kernel(const unsigned short* __restrict__ feats,
             const unsigned short* __restrict__ Wr,
             const float* __restrict__ br,
             float* __restrict__ out) {
    __shared__ unsigned short btile[2][64 * KCHUNK];   // 2 x 8KB, [col][k]

    const int lane = threadIdx.x & 31;
    const int wave = threadIdx.x >> 5;
    const int mb  = blockIdx.x >> 4;    // 256 M blocks (128 rows each)
    const int nst = blockIdx.x & 15;    // 16 N strips of 64 cols
    const int ln = lane & 15;
    const int hi = lane >> 4;
    const int mt = mb * 8 + wave;       // this wave's 16-row tile

    const unsigned short* __restrict__ arow =
        feats + (size_t)(mt * 16 + ln) * FDIM;

    // Async-loader mapping: thread t moves 32B. col = t/4, 16-elem seg = t%4
    const int tcol = threadIdx.x >> 2;
    const int tseg = threadIdx.x & 3;
    const unsigned short* gsrc = Wr + (size_t)(nst * 64 + tcol) * FDIM + tseg * 16;
    unsigned lds0 = (unsigned)(size_t)(&btile[0][tcol * KCHUNK + tseg * 16]);
    unsigned lds1 = (unsigned)(size_t)(&btile[1][tcol * KCHUNK + tseg * 16]);

    v8f acc[4];
    acc[0] = v8f{}; acc[1] = v8f{}; acc[2] = v8f{}; acc[3] = v8f{};

    // Prologue: async-stage chunk 0 into buffer 0
    async_stage32(lds0, gsrc);

    for (int i = 0; i < NKCHUNK; ++i) {
        // own portion of chunk i landed (async ops retire in order per wave)
        asm volatile("s_wait_asynccnt 0x0" ::: "memory");
        __syncthreads();   // all portions landed; all chunk i-1 readers done

        if (i + 1 < NKCHUNK) {
            unsigned lb = ((i + 1) & 1) ? lds1 : lds0;
            async_stage32(lb, gsrc + (i + 1) * KCHUNK);
        }

        const unsigned short* bt = btile[i & 1];
#pragma unroll
        for (int kk = 0; kk < KCHUNK; kk += 32) {
            BF16Frag a;
            a.q[0] = *(const uint4*)(arow + i * KCHUNK + kk + hi * 8);
            a.q[1] = *(const uint4*)(arow + i * KCHUNK + kk + 16 + hi * 8);

#pragma unroll
            for (int nt = 0; nt < 4; ++nt) {
                BF16Frag b;
                const unsigned short* bp = bt + (nt * 16 + ln) * KCHUNK + kk + hi * 8;
                b.q[0] = *(const uint4*)(bp);
                b.q[1] = *(const uint4*)(bp + 16);
                acc[nt] = __builtin_amdgcn_wmma_f32_16x16x32_bf16(
                    false, a.v, false, b.v, (short)0, acc[nt], false, false);
            }
        }
    }

#pragma unroll
    for (int nt = 0; nt < 4; ++nt) {
        const int col = nst * 64 + nt * 16 + ln;
        const float bias = br[col];
#pragma unroll
        for (int r = 0; r < 8; ++r) {
            int row = mt * 16 + hi * 8 + r;
            out[(size_t)row * ODIM + col] = acc[nt][r] + bias;
        }
    }
}

// ---------------------------------------------------------------------------
// Host-side launcher
// ---------------------------------------------------------------------------
extern "C" void kernel_launch(void* const* d_in, const int* in_sizes, int n_in,
                              void* d_out, int out_size, void* d_ws, size_t ws_size,
                              hipStream_t stream) {
    const float* x   = (const float*)d_in[0];   // [16,2048,128]
    const float* ph  = (const float*)d_in[1];   // [16,512]
    const float* We  = (const float*)d_in[2];   // [512,128]
    const float* be  = (const float*)d_in[3];   // [512]
    const float* Wr  = (const float*)d_in[4];   // [1024,2560]
    const float* br  = (const float*)d_in[5];   // [1024]

    float* logits = (float*)d_out;                 // [32768,1024]
    float* hist   = logits + (size_t)MROWS * ODIM; // [32768,512]

    // Workspace layout (bytes):
    //   [0,          131072)    We bf16    (65536)
    //   [131072,    5373952)    Wr bf16    (2621440)
    //   [5373952,  13762560)    x  bf16    (4194304)
    //   [13762560, 80871424)    delta f32  (16777216)
    //   [80871424, 248643584)   feats bf16 (83886080)
    char* ws = (char*)d_ws;
    unsigned short* We16  = (unsigned short*)(ws);
    unsigned short* Wr16  = (unsigned short*)(ws + 131072);
    unsigned short* x16   = (unsigned short*)(ws + 5373952);
    float*          delta = (float*)(ws + 13762560);
    unsigned short* feats = (unsigned short*)(ws + 80871424);

    cvt_bf16_kernel<<<65536 / 256, 256, 0, stream>>>(We, We16, 65536);
    cvt_bf16_kernel<<<2621440 / 256, 256, 0, stream>>>(Wr, Wr16, 2621440);
    cvt_bf16_kernel<<<4194304 / 256, 256, 0, stream>>>(x, x16, 4194304);

    // GEMM1 + wrap: 2048*32 = 65536 tiles, 4 waves/block
    gemm1_delta_kernel<<<16384, 128, 0, stream>>>(x16, We16, be, delta);

    // Phase scan: 8192 chains
    scan_feats_kernel<<<32, 256, 0, stream>>>(delta, ph, hist, feats);

    // GEMM2: 256 M-blocks x 16 N-strips, 8 waves/block
    gemm2_kernel<<<4096, 256, 0, stream>>>(feats, Wr16, br, logits);
}
